// AutoEncoder_83992380441040
// MI455X (gfx1250) — matrix operand order repaired and checked by reading, not compiled, not run on hardware
//
#include <hip/hip_runtime.h>
#include <hip/hip_bf16.h>

#define BNUM   64
#define TMAX   256
#define DIN    256
#define FH     128
#define NCLS   257
#define CHROWS 128
#define YN_STRIDE 264   // 256 + 8 pad (bf16) -> 132 DWORD row stride, conflict-free A fetch
#define WK_STRIDE 136   // 128 + 8 pad (bf16) -> 68 DWORD row stride
#define NTHREADS 256

typedef __bf16 bf16_t;
typedef bf16_t v16bf __attribute__((ext_vector_type(16)));
typedef bf16_t v8bf  __attribute__((ext_vector_type(8)));
typedef float  v8f   __attribute__((ext_vector_type(8)));

union V16 { v16bf v; v8bf h[2]; };

__device__ __forceinline__ float phi_f(float v) {
    // elu(v)+1 : v>0 -> v+1 ; v<=0 -> exp(v)
    return v > 0.0f ? (v + 1.0f) : __expf(v);
}

__device__ __forceinline__ float block_reduce_sum(float v, float* s_red, int tid) {
    s_red[tid] = v;
    __syncthreads();
    #pragma unroll
    for (int s = NTHREADS / 2; s > 0; s >>= 1) {
        if (tid < s) s_red[tid] += s_red[tid + s];
        __syncthreads();
    }
    float r = s_red[0];
    __syncthreads();
    return r;
}

__global__ __launch_bounds__(NTHREADS, 1)
void linattn_autoenc_kernel(const float* __restrict__ x,
                            const int*   __restrict__ batch,
                            const float* __restrict__ Wk,
                            const float* __restrict__ Wq,
                            const float* __restrict__ Wv,
                            const float* __restrict__ ln_g,
                            const float* __restrict__ ln_b,
                            const float* __restrict__ W1,
                            const float* __restrict__ b1,
                            const float* __restrict__ W2,
                            const float* __restrict__ b2,
                            float* __restrict__ out,
                            int total)
{
    __shared__ __align__(16) bf16_t s_wk[DIN * WK_STRIDE];     // Wk as bf16, padded
    __shared__ __align__(16) bf16_t s_yn[CHROWS * YN_STRIDE];  // layernormed chunk, bf16
    __shared__ float s_q[FH];
    __shared__ float s_ylast[DIN];
    __shared__ float s_a[CHROWS];
    __shared__ float s_w[DIN];
    __shared__ float s_lng[DIN];
    __shared__ float s_lnb[DIN];
    __shared__ float s_z[FH];
    __shared__ float s_h[FH];
    __shared__ float s_logits[NCLS];
    __shared__ float s_red[NTHREADS];
    __shared__ int   s_lo, s_n;
    __shared__ float s_den;

    const int tid  = threadIdx.x;
    const int b    = blockIdx.x;
    const int wave = tid >> 5;
    const int lane = tid & 31;

    // ---------- phase 0: segment bounds, stage LN params + Wk(bf16) ----------
    int cnt = 0, lt = 0;
    for (int i = tid; i < total; i += NTHREADS) {
        int v = batch[i];
        cnt += (v == b);
        lt  += (v <  b);
    }
    float fn = block_reduce_sum((float)cnt, s_red, tid);
    float fl = block_reduce_sum((float)lt,  s_red, tid);
    if (tid == 0) {
        int n0 = (int)(fn + 0.5f);
        s_n   = (n0 < 1) ? 1 : n0;
        s_lo  = (int)(fl + 0.5f);
        s_den = 0.0f;
    }
    for (int i = tid; i < DIN; i += NTHREADS) {
        s_lng[i] = ln_g[i];
        s_lnb[i] = ln_b[i];
    }
    #pragma unroll 4
    for (int j = 0; j < (DIN * FH) / NTHREADS; ++j) {
        int idx = j * NTHREADS + tid;
        int r   = idx >> 7;          // /FH
        int cc  = idx & (FH - 1);
        s_wk[r * WK_STRIDE + cc] = (bf16_t)Wk[idx];
    }
    __syncthreads();

    const int n  = s_n;
    const int lo = s_lo;

    // ---------- phase 0b: layernorm of last row, q = phi(yn_last @ Wq) ----------
    {
        float xv   = x[(size_t)(lo + n - 1) * DIN + tid];
        float mu   = block_reduce_sum(xv, s_red, tid) * (1.0f / DIN);
        float dv   = xv - mu;
        float var  = block_reduce_sum(dv * dv, s_red, tid) * (1.0f / DIN);
        float rstd = rsqrtf(var + 1e-5f);
        s_ylast[tid] = dv * rstd * s_lng[tid] + s_lnb[tid];
    }
    __syncthreads();
    if (tid < FH) {
        float acc = 0.0f;
        for (int d = 0; d < DIN; ++d) acc += s_ylast[d] * Wq[d * FH + tid];
        s_q[tid] = phi_f(acc);
    }
    __syncthreads();

    // ---------- main loop over 128-row chunks ----------
    float wacc = 0.0f;                 // this thread owns column d = tid of w
    const int nch = (n + CHROWS - 1) / CHROWS;
    for (int c = 0; c < nch; ++c) {
        __syncthreads();               // protect s_yn reuse across chunks

        // LayerNorm: 16 rows per wave, 8 elements per lane
        for (int rr = 0; rr < 16; ++rr) {
            const int tl = wave * 16 + rr;
            const int t  = c * CHROWS + tl;
            if (t < n) {
                const float* xr = x + (size_t)(lo + t) * DIN + lane * 8;
                float4 p0 = *(const float4*)(xr);
                float4 p1 = *(const float4*)(xr + 4);
                float xv[8] = {p0.x, p0.y, p0.z, p0.w, p1.x, p1.y, p1.z, p1.w};
                float sum = 0.0f;
                #pragma unroll
                for (int j = 0; j < 8; ++j) sum += xv[j];
                #pragma unroll
                for (int off = 16; off >= 1; off >>= 1) sum += __shfl_xor(sum, off, 32);
                float mu = sum * (1.0f / DIN);
                float vs = 0.0f;
                #pragma unroll
                for (int j = 0; j < 8; ++j) { xv[j] -= mu; vs += xv[j] * xv[j]; }
                #pragma unroll
                for (int off = 16; off >= 1; off >>= 1) vs += __shfl_xor(vs, off, 32);
                float rstd = rsqrtf(vs * (1.0f / DIN) + 1e-5f);
                v8bf o;
                #pragma unroll
                for (int j = 0; j < 8; ++j) {
                    int d = lane * 8 + j;
                    o[j] = (bf16_t)(xv[j] * rstd * s_lng[d] + s_lnb[d]);
                }
                *reinterpret_cast<v8bf*>(&s_yn[tl * YN_STRIDE + lane * 8]) = o;
            } else {
                v8bf o;
                #pragma unroll
                for (int j = 0; j < 8; ++j) o[j] = (bf16_t)0.0f;
                *reinterpret_cast<v8bf*>(&s_yn[tl * YN_STRIDE + lane * 8]) = o;
            }
        }
        __syncthreads();

        // K-GEMM tile (16 rows per wave) via v_wmma_f32_16x16x32_bf16,
        // fused with phi() and the dot against q -> a[t].
        // A tiles (8 k-steps) preloaded once; B tiles (8 per n-tile) preloaded
        // as a batch so ds_loads overlap the WMMA chain instead of stalling it.
        {
            const int m0    = wave * 16;
            const int khalf = lane >> 4;                     // K half per ISA A-layout
            const bf16_t* arow = &s_yn[(m0 + (lane & 15)) * YN_STRIDE];

            V16 avs[8];
            #pragma unroll
            for (int kk = 0; kk < 8; ++kk) {
                const int k0 = kk * 32;
                avs[kk].h[0] = *reinterpret_cast<const v8bf*>(arow + k0 + khalf * 8);
                avs[kk].h[1] = *reinterpret_cast<const v8bf*>(arow + k0 + 16 + khalf * 8);
            }

            float aacc[8];
            #pragma unroll
            for (int r = 0; r < 8; ++r) aacc[r] = 0.0f;

            #pragma unroll
            for (int nt = 0; nt < 8; ++nt) {                 // F = 8 n-tiles of 16
                const int n0 = nt * 16;
                // Batch-issue all 8 B-tile ds_loads for this n-tile.
                V16 bvs[8];
                #pragma unroll
                for (int kk = 0; kk < 8; ++kk) {
                    const bf16_t* bcol = &s_wk[(kk * 32 + khalf * 8) * WK_STRIDE + n0 + (lane & 15)];
                    #pragma unroll
                    for (int j = 0; j < 8; ++j) {
                        bvs[kk].v[j]     = bcol[j * WK_STRIDE];
                        bvs[kk].v[8 + j] = bcol[(16 + j) * WK_STRIDE];
                    }
                }
                v8f acc;
                #pragma unroll
                for (int r = 0; r < 8; ++r) acc[r] = 0.0f;
                #pragma unroll
                for (int kk = 0; kk < 8; ++kk) {             // D = 8 k-steps of 32
                    acc = __builtin_amdgcn_wmma_f32_16x16x32_bf16(
                        false, avs[kk].v, false, bvs[kk].v, (short)0, acc, false, false);
                }
                const float qn = s_q[n0 + (lane & 15)];
                #pragma unroll
                for (int r = 0; r < 8; ++r) aacc[r] += phi_f(acc[r]) * qn;
            }
            // row M = r + 8*khalf is spread over 16 lanes (N) -> intra-half reduce
            #pragma unroll
            for (int r = 0; r < 8; ++r) {
                float v = aacc[r];
                #pragma unroll
                for (int off = 1; off < 16; off <<= 1) v += __shfl_xor(v, off, 32);
                const int row_l = m0 + r + 8 * khalf;
                const int t = c * CHROWS + row_l;
                if ((lane & 15) == 0) s_a[row_l] = (t < n) ? v : 0.0f; // mask pad rows
            }
        }
        __syncthreads();

        // w[d] += sum_t a[t] * yn[t][d];  den += sum_t a[t]
        #pragma unroll 4
        for (int tl = 0; tl < CHROWS; ++tl) {
            wacc += s_a[tl] * (float)s_yn[tl * YN_STRIDE + tid];
        }
        {
            float dp  = (tid < CHROWS) ? s_a[tid] : 0.0f;
            float dch = block_reduce_sum(dp, s_red, tid);
            if (tid == 0) s_den += dch;
        }
    }
    __syncthreads();
    s_w[tid] = wacc;
    __syncthreads();

    // ---------- z = (w @ Wv) / (den + eps) ----------
    const float den = s_den + 1e-5f;
    if (tid < FH) {
        float acc = 0.0f;
        for (int d = 0; d < DIN; ++d) acc += s_w[d] * Wv[d * FH + tid];
        float zv = acc / den;
        s_z[tid] = zv;
        out[(size_t)b * FH + tid] = zv;
    }
    __syncthreads();

    // ---------- decoder MLP ----------
    if (tid < FH) {
        float acc = b1[tid];
        for (int l = 0; l < FH; ++l) acc += s_z[l] * W1[l * FH + tid];
        s_h[tid] = tanhf(acc);
    }
    __syncthreads();
    for (int cix = tid; cix < NCLS; cix += NTHREADS) {
        float acc = b2[cix];
        for (int j = 0; j < FH; ++j) acc += s_h[j] * W2[j * NCLS + cix];
        s_logits[cix] = acc;
        out[(size_t)BNUM * FH + (size_t)b * NCLS + cix] = acc;
    }
    __syncthreads();
    if (tid == 0) {
        int best = 0;
        float bvv = s_logits[0];
        for (int i2 = 1; i2 < NCLS; ++i2) {
            if (s_logits[i2] > bvv) { bvv = s_logits[i2]; best = i2; }
        }
        out[(size_t)BNUM * FH + (size_t)BNUM * NCLS + b] = (float)best;
    }
}

extern "C" void kernel_launch(void* const* d_in, const int* in_sizes, int n_in,
                              void* d_out, int out_size, void* d_ws, size_t ws_size,
                              hipStream_t stream) {
    const float* x     = (const float*)d_in[0];
    const int*   batch = (const int*)  d_in[1];
    const float* Wk    = (const float*)d_in[2];
    const float* Wq    = (const float*)d_in[3];
    const float* Wv    = (const float*)d_in[4];
    const float* lng   = (const float*)d_in[5];
    const float* lnb   = (const float*)d_in[6];
    const float* W1    = (const float*)d_in[7];
    const float* b1    = (const float*)d_in[8];
    const float* W2    = (const float*)d_in[9];
    const float* b2    = (const float*)d_in[10];
    float* out = (float*)d_out;
    const int total = in_sizes[1];   // length of batch array = B*T
    (void)n_in; (void)out_size; (void)d_ws; (void)ws_size;

    linattn_autoenc_kernel<<<dim3(BNUM), dim3(NTHREADS), 0, stream>>>(
        x, batch, Wk, Wq, Wv, lng, lnb, W1, b1, W2, b2, out, total);
}
